// MultiHeadAttention_34325378629898
// MI455X (gfx1250) — matrix operand order repaired and checked by reading, not compiled, not run on hardware
//
#include <hip/hip_runtime.h>
#include <hip/hip_bf16.h>

// ---------------------------------------------------------------------------
// MultiHeadAttention + residual + LayerNorm for gfx1250 (MI455X, wave32, WMMA)
// B=4 S=2048 H=1024 NH=16 HD=64
// ---------------------------------------------------------------------------

typedef _Float16 half8 __attribute__((ext_vector_type(8)));
typedef _Float16 v16h  __attribute__((ext_vector_type(16)));
typedef float    v8f   __attribute__((ext_vector_type(8)));

#define WMMA_F16(a, b, c) \
    __builtin_amdgcn_wmma_f32_16x16x32_f16(false, (a), false, (b), (short)0, (c), false, false)

static constexpr int BB  = 4;
static constexpr int SS  = 2048;
static constexpr int HH  = 1024;
static constexpr int NHD = 16;
static constexpr int HD  = 64;
static constexpr int MM  = BB * SS;   // 8192 rows

// ---------------------------------------------------------------------------
// Fragment loaders (wave32 WMMA layouts per CDNA5 ISA 7.12.2)
// A (16x32 f16): lane<16 -> row=lane, K chunks [k0..k0+7] and [k0+16..k0+23];
//                lane>=16 -> same row set, chunks shifted by +8.
// B (32x16 f16): lane<16 -> col=lane, K chunks [k0..k0+7] and [k0+16..k0+23]; lane>=16 -> chunks shifted by +8.
// ---------------------------------------------------------------------------
__device__ __forceinline__ v16h load_frag_a(const _Float16* p, int ld, int m0,
                                            int k0, int lane) {
    int r  = lane & 15;
    int hi = (lane >> 4) & 1;
    const _Float16* base = p + (size_t)(m0 + r) * (size_t)ld + k0 + hi * 8;
    half8 lo = *(const half8*)(base);
    half8 h2 = *(const half8*)(base + 16);
    return __builtin_shufflevector(lo, h2, 0, 1, 2, 3, 4, 5, 6, 7,
                                   8, 9, 10, 11, 12, 13, 14, 15);
}

__device__ __forceinline__ v16h load_frag_b(const _Float16* p, int ld, int srcrow,
                                            int k0, int hi) {
    const _Float16* base = p + (size_t)srcrow * (size_t)ld + k0 + hi * 8;
    half8 lo = *(const half8*)(base);
    half8 h2 = *(const half8*)(base + 16);
    return __builtin_shufflevector(lo, h2, 0, 1, 2, 3, 4, 5, 6, 7,
                                   8, 9, 10, 11, 12, 13, 14, 15);
}

// ---------------------------------------------------------------------------
// f32 -> f16 conversion (grid-stride)
// ---------------------------------------------------------------------------
__global__ __launch_bounds__(256) void k_cvt(const float* __restrict__ src,
                                             _Float16* __restrict__ dst, int n) {
    for (int i = blockIdx.x * blockDim.x + threadIdx.x; i < n;
         i += gridDim.x * blockDim.x)
        dst[i] = (_Float16)src[i];
}

// ---------------------------------------------------------------------------
// GEMM: out = A(f16, MxK row-major) @ W(f16, NxK row-major)^T + bias
// One wave computes a 32x32 output tile, 4 WMMAs per K-step of 32.
// mode 0: f16 row-major [M,H]     (Q / K caches)
// mode 1: f16 transposed-per-head Vt[b][h][d][s]  (V cache)
// mode 2: f32 row-major [M,H]     (output projection)
// ---------------------------------------------------------------------------
__global__ __launch_bounds__(256) void k_gemm(const _Float16* __restrict__ A,
                                              const _Float16* __restrict__ W,
                                              const float* __restrict__ bias,
                                              _Float16* __restrict__ out16,
                                              float* __restrict__ out32,
                                              int mode) {
    const int lane = threadIdx.x & 31;
    const int gw   = (blockIdx.x << 3) + (threadIdx.x >> 5);
    const int tn   = gw & 31;   // 1024/32 = 32 N tiles
    const int tm   = gw >> 5;   // 8192/32 = 256 M tiles
    const int m0   = tm << 5;
    const int n0   = tn << 5;
    const int r    = lane & 15;
    const int hi   = (lane >> 4) & 1;

    v8f c00 = {}, c01 = {}, c10 = {}, c11 = {};
    for (int k0 = 0; k0 < HH; k0 += 32) {
        v16h a0 = load_frag_a(A, HH, m0, k0, lane);
        v16h a1 = load_frag_a(A, HH, m0 + 16, k0, lane);
        v16h b0 = load_frag_b(W, HH, n0 + r, k0, hi);
        v16h b1 = load_frag_b(W, HH, n0 + 16 + r, k0, hi);
        c00 = WMMA_F16(a0, b0, c00);
        c01 = WMMA_F16(a0, b1, c01);
        c10 = WMMA_F16(a1, b0, c10);
        c11 = WMMA_F16(a1, b1, c11);
    }

    const float bb0 = bias[n0 + r];
    const float bb1 = bias[n0 + 16 + r];
#pragma unroll
    for (int g = 0; g < 8; ++g) {
        c00[g] += bb0; c01[g] += bb1; c10[g] += bb0; c11[g] += bb1;
    }

    auto store_tile = [&](v8f c, int mrow0, int ncol0) {
#pragma unroll
        for (int g = 0; g < 8; ++g) {
            int row = mrow0 + g + hi * 8;
            int col = ncol0 + r;
            float val = c[g];
            if (mode == 0) {
                out16[(size_t)row * HH + col] = (_Float16)val;
            } else if (mode == 1) {
                int h = col >> 6, d = col & 63;
                int b = row >> 11, s = row & (SS - 1);
                out16[(((size_t)b * NHD + h) * HD + d) * SS + s] = (_Float16)val;
            } else {
                out32[(size_t)row * HH + col] = val;
            }
        }
    };
    store_tile(c00, m0, n0);
    store_tile(c01, m0, n0 + 16);
    store_tile(c10, m0 + 16, n0);
    store_tile(c11, m0 + 16, n0 + 16);
}

// ---------------------------------------------------------------------------
// Flash attention: one wave handles 16 query rows of one head.
// Online softmax over key tiles of 32; P restaged via LDS into A-fragment.
// ---------------------------------------------------------------------------
__global__ __launch_bounds__(256) void k_attn(const _Float16* __restrict__ Q,
                                              const _Float16* __restrict__ Kc,
                                              const _Float16* __restrict__ Vt,
                                              const int* __restrict__ mask,
                                              _Float16* __restrict__ Att) {
    __shared__ __align__(16) _Float16 Plds[8][16 * 32];

    const int tid  = threadIdx.x;
    const int lane = tid & 31;
    const int wav  = tid >> 5;
    const int gw   = (blockIdx.x << 3) + wav;   // 8192 waves total
    const int qt = gw & 127;           // S/16 = 128 query tiles
    const int h  = (gw >> 7) & 15;     // head
    const int b  = gw >> 11;           // batch
    const int q0 = qt << 4;
    const int r  = lane & 15;
    const int hi = (lane >> 4) & 1;
    const float LOG2E = 1.4426950408889634f;

    // Q tile fragments (16 rows x 64 dims = two K=32 A-frags), kept resident
    const v16h aQ0 = load_frag_a(Q, HH, b * SS + q0, h * HD + 0, lane);
    const v16h aQ1 = load_frag_a(Q, HH, b * SS + q0, h * HD + 32, lane);

    v8f o0 = {}, o1 = {}, o2 = {}, o3 = {};
    float mrow[8], lrow[8];
#pragma unroll
    for (int g = 0; g < 8; ++g) { mrow[g] = -3.0e38f; lrow[g] = 0.0f; }

    _Float16* Pw = &Plds[wav][0];
    const int vrow0 = (b * NHD + h) * HD;

    for (int jb = 0; jb < SS; jb += 32) {
        const int mk0 = mask[b * SS + jb + r];
        const int mk1 = mask[b * SS + jb + 16 + r];

        // scores: S_tile[16x32] = Q(16x64) . K_tile(32x64)^T
        v16h bK00 = load_frag_b(Kc, HH, b * SS + jb + r,      h * HD + 0,  hi);
        v16h bK01 = load_frag_b(Kc, HH, b * SS + jb + r,      h * HD + 32, hi);
        v16h bK10 = load_frag_b(Kc, HH, b * SS + jb + 16 + r, h * HD + 0,  hi);
        v16h bK11 = load_frag_b(Kc, HH, b * SS + jb + 16 + r, h * HD + 32, hi);
        v8f s0 = {}, s1 = {};
        s0 = WMMA_F16(aQ0, bK00, s0);
        s0 = WMMA_F16(aQ1, bK01, s0);
        s1 = WMMA_F16(aQ0, bK10, s1);
        s1 = WMMA_F16(aQ1, bK11, s1);

        // online softmax update (per row = VGPR index g within half-wave)
        float corr[8];
#pragma unroll
        for (int g = 0; g < 8; ++g) {
            float x0 = s0[g] * 0.125f; if (mk0 == 0) x0 = -1.0e9f;
            float x1 = s1[g] * 0.125f; if (mk1 == 0) x1 = -1.0e9f;
            float tmax = fmaxf(x0, x1);
#pragma unroll
            for (int off = 1; off < 16; off <<= 1)
                tmax = fmaxf(tmax, __shfl_xor(tmax, off, 32));
            float mn = fmaxf(mrow[g], tmax);
            corr[g] = exp2f((mrow[g] - mn) * LOG2E);
            float p0 = exp2f((x0 - mn) * LOG2E);
            float p1 = exp2f((x1 - mn) * LOG2E);
            float rs = p0 + p1;
#pragma unroll
            for (int off = 1; off < 16; off <<= 1)
                rs += __shfl_xor(rs, off, 32);
            lrow[g] = lrow[g] * corr[g] + rs;
            mrow[g] = mn;
            int prow = g + hi * 8;
            Pw[prow * 32 + r]      = (_Float16)p0;
            Pw[prow * 32 + 16 + r] = (_Float16)p1;
        }
#pragma unroll
        for (int g = 0; g < 8; ++g) {
            o0[g] *= corr[g]; o1[g] *= corr[g];
            o2[g] *= corr[g]; o3[g] *= corr[g];
        }

        // O += P(16x32) . V_tile(32x64); Vt rows are contiguous along keys
        v16h aP  = load_frag_a(Pw, 32, 0, 0, lane);
        v16h bV0 = load_frag_b(Vt, SS, vrow0 + 0  + r, jb, hi);
        v16h bV1 = load_frag_b(Vt, SS, vrow0 + 16 + r, jb, hi);
        v16h bV2 = load_frag_b(Vt, SS, vrow0 + 32 + r, jb, hi);
        v16h bV3 = load_frag_b(Vt, SS, vrow0 + 48 + r, jb, hi);
        o0 = WMMA_F16(aP, bV0, o0);
        o1 = WMMA_F16(aP, bV1, o1);
        o2 = WMMA_F16(aP, bV2, o2);
        o3 = WMMA_F16(aP, bV3, o3);
    }

#pragma unroll
    for (int g = 0; g < 8; ++g) {
        int srow = q0 + g + hi * 8;
        float inv = 1.0f / lrow[g];
        size_t base = ((size_t)(b * SS + srow)) * HH + h * HD;
        Att[base + 0  + r] = (_Float16)(o0[g] * inv);
        Att[base + 16 + r] = (_Float16)(o1[g] * inv);
        Att[base + 32 + r] = (_Float16)(o2[g] * inv);
        Att[base + 48 + r] = (_Float16)(o3[g] * inv);
    }
}

// ---------------------------------------------------------------------------
// Residual + LayerNorm: one block per row of 1024
// ---------------------------------------------------------------------------
__global__ __launch_bounds__(256) void k_ln(const float* __restrict__ x,
                                            const float* __restrict__ proj,
                                            const float* __restrict__ gamma,
                                            const float* __restrict__ beta,
                                            float* __restrict__ out) {
    __shared__ float red[256];
    const int row = blockIdx.x;
    const int tid = threadIdx.x;

    float v[4];
    float s = 0.0f, s2 = 0.0f;
#pragma unroll
    for (int i = 0; i < 4; ++i) {
        int c = tid + i * 256;
        float t = x[(size_t)row * HH + c] + proj[(size_t)row * HH + c];
        v[i] = t;
        s += t;
        s2 += t * t;
    }
    red[tid] = s;
    __syncthreads();
    for (int off = 128; off > 0; off >>= 1) {
        if (tid < off) red[tid] += red[tid + off];
        __syncthreads();
    }
    float mean = red[0] * (1.0f / HH);
    __syncthreads();
    red[tid] = s2;
    __syncthreads();
    for (int off = 128; off > 0; off >>= 1) {
        if (tid < off) red[tid] += red[tid + off];
        __syncthreads();
    }
    float var = red[0] * (1.0f / HH) - mean * mean;
    float rstd = rsqrtf(var + 1e-5f);
#pragma unroll
    for (int i = 0; i < 4; ++i) {
        int c = tid + i * 256;
        out[(size_t)row * HH + c] = (v[i] - mean) * rstd * gamma[c] + beta[c];
    }
}

// ---------------------------------------------------------------------------
// Launch
// ---------------------------------------------------------------------------
extern "C" void kernel_launch(void* const* d_in, const int* in_sizes, int n_in,
                              void* d_out, int out_size, void* d_ws, size_t ws_size,
                              hipStream_t stream) {
    const float* x     = (const float*)d_in[0];
    const int*   mask  = (const int*)d_in[1];
    const float* Wq    = (const float*)d_in[2];
    const float* bq    = (const float*)d_in[3];
    const float* Wk    = (const float*)d_in[4];
    const float* bk    = (const float*)d_in[5];
    const float* Wv    = (const float*)d_in[6];
    const float* bv    = (const float*)d_in[7];
    const float* Wo    = (const float*)d_in[8];
    const float* bo    = (const float*)d_in[9];
    const float* gamma = (const float*)d_in[10];
    const float* beta  = (const float*)d_in[11];

    char* ws = (char*)d_ws;
    const size_t szXH = (size_t)MM * HH * sizeof(_Float16);   // 16 MB
    const size_t szWH = (size_t)HH * HH * sizeof(_Float16);   //  2 MB
    _Float16* xh  = (_Float16*)(ws);
    _Float16* wqh = (_Float16*)(ws + szXH);
    _Float16* wkh = (_Float16*)(ws + szXH + szWH);
    _Float16* wvh = (_Float16*)(ws + szXH + 2 * szWH);
    _Float16* woh = (_Float16*)(ws + szXH + 3 * szWH);
    _Float16* qh  = (_Float16*)(ws + szXH + 4 * szWH);
    _Float16* kh  = (_Float16*)(ws + 2 * szXH + 4 * szWH);
    _Float16* vth = (_Float16*)(ws + 3 * szXH + 4 * szWH);
    _Float16* ath = (_Float16*)(ws + 4 * szXH + 4 * szWH);
    float*    prj = (float*)   (ws + 5 * szXH + 4 * szWH);    // 32 MB f32
    float*    out = (float*)d_out;

    // 1) convert inputs to f16
    k_cvt<<<512, 256, 0, stream>>>(x,  xh,  MM * HH);
    k_cvt<<<256, 256, 0, stream>>>(Wq, wqh, HH * HH);
    k_cvt<<<256, 256, 0, stream>>>(Wk, wkh, HH * HH);
    k_cvt<<<256, 256, 0, stream>>>(Wv, wvh, HH * HH);
    k_cvt<<<256, 256, 0, stream>>>(Wo, woh, HH * HH);

    // 2) QKV projections (8192 waves = 1024 blocks x 8 waves each)
    k_gemm<<<1024, 256, 0, stream>>>(xh, wqh, bq, qh,  nullptr, 0);
    k_gemm<<<1024, 256, 0, stream>>>(xh, wkh, bk, kh,  nullptr, 0);
    k_gemm<<<1024, 256, 0, stream>>>(xh, wvh, bv, vth, nullptr, 1);

    // 3) flash attention (B*NH*(S/16) = 8192 waves)
    k_attn<<<1024, 256, 0, stream>>>(qh, kh, vth, mask, ath);

    // 4) output projection -> f32
    k_gemm<<<1024, 256, 0, stream>>>(ath, woh, bo, nullptr, prj, 2);

    // 5) residual + LayerNorm
    k_ln<<<MM, 256, 0, stream>>>(x, prj, gamma, beta, out);
}
